// GATLayer_12206297055519
// MI455X (gfx1250) — compile-verified, matrix operand order checked by reading, not run on hardware
//
#include <hip/hip_runtime.h>

#define N_NODES 50000
#define N_EDGES 800000
#define IN_DIM 256
#define OUT_DIM 128
#define NEG_SLOPE 0.01f

typedef __attribute__((ext_vector_type(16))) __bf16 v16bf;
typedef __attribute__((ext_vector_type(8)))  float  v8f;

// W1 bf16 fragment buffer: [ntile(8)][kstep(8)][lane(32)][elem(16)]
#define W1BF_GROUPS (8 * 8 * 32)

// ---------------------------------------------------------------------------
// Kernel 0: one-time repack of W1 (f32, KxN row-major) into bf16 per-lane
// WMMA B-fragment layout so the GEMM loads B as contiguous 32B per lane.
// ---------------------------------------------------------------------------
__global__ void w1_prep(const float* __restrict__ W1, __bf16* __restrict__ w1bf) {
  const int g = blockIdx.x * blockDim.x + threadIdx.x;   // fragment group id
  if (g >= W1BF_GROUPS) return;
  const int lane = g & 31;
  const int ks   = (g >> 5) & 7;
  const int nt   = g >> 8;
  const int col  = nt * 16 + (lane & 15);
  const int kb   = ks * 32 + ((lane >> 4) * 8);
  __bf16* o = w1bf + (size_t)g * 16;
#pragma unroll
  for (int j = 0; j < 8; ++j)
    o[j] = (__bf16)W1[(size_t)(kb + j) * OUT_DIM + col];
#pragma unroll
  for (int j = 0; j < 8; ++j)
    o[8 + j] = (__bf16)W1[(size_t)(kb + 16 + j) * OUT_DIM + col];
}

// ---------------------------------------------------------------------------
// Kernel 1: z = h @ W1 + b1 via v_wmma_f32_16x16x32_bf16
// grid.x = M/16 = 3125 blocks, 256 threads = 8 waves, wave w owns N-tile w.
// A-strip (16 x 256 f32 = 16KB) staged into LDS with async global->LDS loads.
// ---------------------------------------------------------------------------
__global__ void gemm_z_wmma(const float* __restrict__ h,
                            const __bf16* __restrict__ w1bf,
                            const float* __restrict__ b1,
                            float* __restrict__ z) {
  __shared__ __align__(16) float lds_a[16 * IN_DIM];   // 16 KB

  const int tid  = threadIdx.x;
  const int lane = tid & 31;
  const int wave = tid >> 5;            // 0..7 -> N tile
  const int m0   = blockIdx.x * 16;
  const int lmod = lane & 15;
  const int khi  = (lane >> 4) * 8;     // 0 (lanes 0-15) or 8 (lanes 16-31)

  // --- async stage: 1024 x 16B chunks, 4 per thread, tracked by ASYNCcnt ---
#pragma unroll
  for (int i = 0; i < 4; ++i) {
    const int c = tid + i * 256;                        // 16-byte chunk index
    const unsigned lds_addr = (unsigned)(size_t)&lds_a[c * 4];
    const float* gsrc = h + (size_t)m0 * IN_DIM + c * 4;
    asm volatile("global_load_async_to_lds_b128 %0, %1, off"
                 :: "v"(lds_addr), "v"(gsrc) : "memory");
  }
  asm volatile("s_wait_asynccnt 0" ::: "memory");
  __syncthreads();

  const float* arow = lds_a + (size_t)lmod * IN_DIM;    // this lane's A row
  const __bf16* bfrag = w1bf + (size_t)wave * 8 * 32 * 16 + (size_t)lane * 16;

  v8f acc = {};
#pragma unroll
  for (int ks = 0; ks < 8; ++ks) {
    const int kb = ks * 32 + khi;
    // A fragment from LDS: K = kb..kb+7 (elems 0-7), kb+16..kb+23 (8-15)
    const float4 a0 = *(const float4*)(arow + kb);
    const float4 a1 = *(const float4*)(arow + kb + 4);
    const float4 a2 = *(const float4*)(arow + kb + 16);
    const float4 a3 = *(const float4*)(arow + kb + 20);
    v16bf a;
    a[0]=(__bf16)a0.x; a[1]=(__bf16)a0.y; a[2]=(__bf16)a0.z; a[3]=(__bf16)a0.w;
    a[4]=(__bf16)a1.x; a[5]=(__bf16)a1.y; a[6]=(__bf16)a1.z; a[7]=(__bf16)a1.w;
    a[8]=(__bf16)a2.x; a[9]=(__bf16)a2.y; a[10]=(__bf16)a2.z; a[11]=(__bf16)a2.w;
    a[12]=(__bf16)a3.x; a[13]=(__bf16)a3.y; a[14]=(__bf16)a3.z; a[15]=(__bf16)a3.w;

    // B fragment: contiguous 32B per lane (preconverted layout)
    const v16bf b = *(const v16bf*)(bfrag + (size_t)ks * 32 * 16);

    acc = __builtin_amdgcn_wmma_f32_16x16x32_bf16(
        /*neg_a=*/false, a, /*neg_b=*/false, b,
        /*c_mod=*/(short)0, acc, /*reuse_a=*/false, /*reuse_b=*/false);
  }

  // C/D layout: VGPR i -> M = i + (lane>=16 ? 8 : 0), N = lane%16
  const int bcol  = wave * 16 + lmod;
  const int mbase = m0 + khi;
  const float bias = b1[bcol];
#pragma unroll
  for (int i = 0; i < 8; ++i)
    z[(size_t)(mbase + i) * OUT_DIM + bcol] = acc[i] + bias;
}

// ---------------------------------------------------------------------------
// order-preserving float <-> uint encoding for atomicMax on floats
// ---------------------------------------------------------------------------
__device__ __forceinline__ unsigned enc_f(float x) {
  unsigned u = __float_as_uint(x);
  return (u & 0x80000000u) ? ~u : (u | 0x80000000u);
}
__device__ __forceinline__ float dec_f(unsigned u) {
  return __uint_as_float((u & 0x80000000u) ? (u & 0x7FFFFFFFu) : ~u);
}

// ---------------------------------------------------------------------------
// Kernel 2: per-node scores + init of reduction buffers + zero h_out
// ---------------------------------------------------------------------------
__global__ void node_scores_init(const float* __restrict__ z,
                                 const float* __restrict__ a_w,
                                 float* __restrict__ s_src,
                                 float* __restrict__ s_dst,
                                 unsigned* __restrict__ e_max_enc,
                                 float* __restrict__ denom,
                                 float* __restrict__ h_out) {
  const int node = blockIdx.x * 8 + (threadIdx.x >> 5);
  const int lane = threadIdx.x & 31;
  if (node >= N_NODES) return;

  const float4 zv = *(const float4*)(z + (size_t)node * OUT_DIM + lane * 4);
  const float4 as = *(const float4*)(a_w + lane * 4);
  const float4 ad = *(const float4*)(a_w + OUT_DIM + lane * 4);
  float ss = zv.x * as.x + zv.y * as.y + zv.z * as.z + zv.w * as.w;
  float sd = zv.x * ad.x + zv.y * ad.y + zv.z * ad.z + zv.w * ad.w;
#pragma unroll
  for (int off = 16; off > 0; off >>= 1) {
    ss += __shfl_down(ss, off, 32);
    sd += __shfl_down(sd, off, 32);
  }

  *(float4*)(h_out + (size_t)node * OUT_DIM + lane * 4) =
      make_float4(0.f, 0.f, 0.f, 0.f);

  if (lane == 0) {
    s_src[node] = ss;
    s_dst[node] = sd;
    e_max_enc[node] = 0x007FFFFFu;  // enc(-inf)
    denom[node] = 0.f;
  }
}

// ---------------------------------------------------------------------------
// Kernel 3: edge logits + segment max (atomicMax on encoded uint)
// ---------------------------------------------------------------------------
__global__ void edge_max(const int* __restrict__ src,
                         const int* __restrict__ dst,
                         const float* __restrict__ s_src,
                         const float* __restrict__ s_dst,
                         const float* __restrict__ a_b,
                         float* __restrict__ e_buf,
                         unsigned* __restrict__ e_max_enc) {
  const int e = blockIdx.x * blockDim.x + threadIdx.x;
  if (e >= N_EDGES) return;
  const int d = dst[e];
  float v = s_src[src[e]] + s_dst[d] + a_b[0];
  v = (v > 0.f) ? v : NEG_SLOPE * v;   // leaky_relu
  e_buf[e] = v;
  atomicMax(&e_max_enc[d], enc_f(v));
}

// ---------------------------------------------------------------------------
// Kernel 4: exp(e - max[dst]) + segment sum
// ---------------------------------------------------------------------------
__global__ void edge_expsum(const int* __restrict__ dst,
                            const unsigned* __restrict__ e_max_enc,
                            float* __restrict__ e_buf,
                            float* __restrict__ denom) {
  const int e = blockIdx.x * blockDim.x + threadIdx.x;
  if (e >= N_EDGES) return;
  const int d = dst[e];
  const float ex = __expf(e_buf[e] - dec_f(e_max_enc[d]));
  e_buf[e] = ex;
  atomicAdd(&denom[d], ex);
}

// ---------------------------------------------------------------------------
// Kernel 5: weighted scatter  h_out[dst] += alpha * z[src]
// one wave32 per edge, lane handles 4 features -> 4x global_atomic_add_f32
// ---------------------------------------------------------------------------
__global__ void edge_scatter(const int* __restrict__ src,
                             const int* __restrict__ dst,
                             const float* __restrict__ z,
                             const float* __restrict__ e_buf,
                             const float* __restrict__ denom,
                             float* __restrict__ h_out) {
  const long long tid = (long long)blockIdx.x * blockDim.x + threadIdx.x;
  const int e = (int)(tid >> 5);
  const int lane = threadIdx.x & 31;
  if (e >= N_EDGES) return;
  const int s = src[e], d = dst[e];
  const float alpha = e_buf[e] / denom[d];
  const float4 zv = *(const float4*)(z + (size_t)s * OUT_DIM + lane * 4);
  float* o = h_out + (size_t)d * OUT_DIM + lane * 4;
  atomicAdd(o + 0, alpha * zv.x);
  atomicAdd(o + 1, alpha * zv.y);
  atomicAdd(o + 2, alpha * zv.z);
  atomicAdd(o + 3, alpha * zv.w);
}

// ---------------------------------------------------------------------------
extern "C" void kernel_launch(void* const* d_in, const int* in_sizes, int n_in,
                              void* d_out, int out_size, void* d_ws, size_t ws_size,
                              hipStream_t stream) {
  const float* h   = (const float*)d_in[0];
  const float* W1  = (const float*)d_in[1];
  const float* b1  = (const float*)d_in[2];
  const float* a_w = (const float*)d_in[3];
  const float* a_b = (const float*)d_in[4];
  const int*   src = (const int*)d_in[5];
  const int*   dst = (const int*)d_in[6];
  float* h_out = (float*)d_out;

  // workspace layout: w1bf (64KB, 32B-aligned) | z | s_src | s_dst |
  //                   e_max_enc | denom | e_buf
  __bf16* w1bf = (__bf16*)d_ws;                              // 32768 bf16
  float* z = (float*)d_ws + 16384;                           // 6,400,000
  float* s_src = z + (size_t)N_NODES * OUT_DIM;              //    50,000
  float* s_dst = s_src + N_NODES;                            //    50,000
  unsigned* e_max_enc = (unsigned*)(s_dst + N_NODES);        //    50,000
  float* denom = (float*)(e_max_enc + N_NODES);              //    50,000
  float* e_buf = denom + N_NODES;                            //   800,000

  w1_prep<<<(W1BF_GROUPS + 255) / 256, 256, 0, stream>>>(W1, w1bf);

  gemm_z_wmma<<<N_NODES / 16, 256, 0, stream>>>(h, w1bf, b1, z);

  node_scores_init<<<(N_NODES + 7) / 8, 256, 0, stream>>>(
      z, a_w, s_src, s_dst, e_max_enc, denom, h_out);

  edge_max<<<(N_EDGES + 255) / 256, 256, 0, stream>>>(
      src, dst, s_src, s_dst, a_b, e_buf, e_max_enc);

  edge_expsum<<<(N_EDGES + 255) / 256, 256, 0, stream>>>(
      dst, e_max_enc, e_buf, denom);

  edge_scatter<<<((long long)N_EDGES * 32 + 255) / 256, 256, 0, stream>>>(
      src, dst, z, e_buf, denom, h_out);
}
